// Transformer_17729624997901
// MI455X (gfx1250) — matrix operand order repaired and checked by reading, not compiled, not run on hardware
//
#include <hip/hip_runtime.h>
#include <math.h>

// ---------------- problem constants ----------------
constexpr int NL   = 6;
constexpr int DIM  = 256;
constexpr int NHD  = 8;       // heads
constexpr int HD   = 32;      // head dim
constexpr int DFF  = 2048;
constexpr int LVN  = 4;       // levels
constexpr int PN   = 8;       // points
constexpr int NQ   = 1000;
constexpr int NQP  = 1024;    // padded query count (tile aligned)
constexpr int BB   = 4;
constexpr int TOPK = 100;
constexpr int STOT = 64*64 + 32*32 + 16*16 + 8*8;  // 5440
constexpr float SCALE = 0.17677669529663687f;      // 1/sqrt(32)
constexpr float NEGBIG = -1.0e9f;

typedef __attribute__((ext_vector_type(16))) _Float16 v16h;
typedef __attribute__((ext_vector_type(8)))  _Float16 v8h;
typedef __attribute__((ext_vector_type(8)))  float    v8f;

#define DEV __device__ __forceinline__

// ---------- WMMA fragment helpers (CDNA5 16x16x32 f16 layouts) ----------
// A (16x32): lane L: m=L&15, h=L>>4 ; elem e: k = h*16 + e
// B (32x16): lane L: n=L&15, g=L>>4 ; elem e: k = g*16 + e
// C/D (16x16 f32): lane L: n=L&15 ; elem r: m = r + 8*(L>>4)

// pack 16 f32 (two aligned 8-float runs) into an f16 A-fragment
DEV v16h pack16(const float* __restrict__ p, const float* __restrict__ p2) {
  float4 x0 = ((const float4*)p)[0];
  float4 x1 = ((const float4*)p)[1];
  float4 x2 = ((const float4*)p2)[0];
  float4 x3 = ((const float4*)p2)[1];
  v16h r;
  r[0]=(_Float16)x0.x; r[1]=(_Float16)x0.y; r[2]=(_Float16)x0.z; r[3]=(_Float16)x0.w;
  r[4]=(_Float16)x1.x; r[5]=(_Float16)x1.y; r[6]=(_Float16)x1.z; r[7]=(_Float16)x1.w;
  r[8]=(_Float16)x2.x; r[9]=(_Float16)x2.y; r[10]=(_Float16)x2.z; r[11]=(_Float16)x2.w;
  r[12]=(_Float16)x3.x; r[13]=(_Float16)x3.y; r[14]=(_Float16)x3.z; r[15]=(_Float16)x3.w;
  return r;
}
// concatenate two aligned 8-half runs into a fragment
DEV v16h cat8(const _Float16* __restrict__ p, const _Float16* __restrict__ p2) {
  v8h a = *(const v8h*)p;
  v8h b = *(const v8h*)p2;
  v16h r;
#pragma unroll
  for (int e = 0; e < 8; ++e) { r[e] = a[e]; r[8 + e] = b[e]; }
  return r;
}

DEV v8f wmma_f16(v16h a, v16h b, v8f c) {
  return __builtin_amdgcn_wmma_f32_16x16x32_f16(false, a, false, b,
                                                (short)0, c, false, false);
}

// ---------------- generic GEMM: C = act(A(MxK,f32) @ Wt(f16, N-major) + bias [+resid])
// N,K multiples of 32; M multiple of 32. 32x32 tile per wave (4 WMMA / K-step).
template<int N, int K>
__global__ void __launch_bounds__(128)
gemm_nn(const float* __restrict__ A, const _Float16* __restrict__ Wt,
        const float* __restrict__ bias, const float* __restrict__ resid,
        float* __restrict__ C, int M, int flags) {
  int wave = threadIdx.x >> 5;
  int lane = threadIdx.x & 31;
  int row0 = (blockIdx.y * 4 + wave) * 32;
  int col0 = blockIdx.x * 32;
  if (row0 >= M) return;                 // wave-uniform
  int m = lane & 15, g = lane >> 4;
  const float* arow0 = A + (size_t)(row0 + m) * K + (g << 4);
  const float* arow1 = arow0 + (size_t)16 * K;
  const _Float16* bcol0 = Wt + (size_t)(col0 + m) * K + (g << 4);
  const _Float16* bcol1 = bcol0 + (size_t)16 * K;
  v8f acc00 = {}, acc01 = {}, acc10 = {}, acc11 = {};
  for (int k0 = 0; k0 < K; k0 += 32) {
    if (k0 + 32 < K) {
      __builtin_prefetch(arow0 + k0 + 32, 0, 1);
      __builtin_prefetch(bcol0 + k0 + 32, 0, 1);
    }
    v16h af0 = pack16(arow0 + k0, arow0 + k0 + 8);
    v16h af1 = pack16(arow1 + k0, arow1 + k0 + 8);
    v16h bf0 = *(const v16h*)(bcol0 + k0);
    v16h bf1 = *(const v16h*)(bcol1 + k0);
    acc00 = wmma_f16(af0, bf0, acc00);
    acc01 = wmma_f16(af0, bf1, acc01);
    acc10 = wmma_f16(af1, bf0, acc10);
    acc11 = wmma_f16(af1, bf1, acc11);
  }
  int cn = col0 + m;
  float bv0 = bias ? bias[cn] : 0.0f;
  float bv1 = bias ? bias[cn + 16] : 0.0f;
#pragma unroll
  for (int r = 0; r < 8; ++r) {
    int cm0 = row0 + r + (g << 3);
    int cm1 = cm0 + 16;
    size_t o0 = (size_t)cm0 * N + cn;
    size_t o1 = (size_t)cm1 * N + cn;
    float v00 = acc00[r] + bv0, v01 = acc01[r] + bv1;
    float v10 = acc10[r] + bv0, v11 = acc11[r] + bv1;
    if (resid) {
      v00 += resid[o0]; v01 += resid[o0 + 16];
      v10 += resid[o1]; v11 += resid[o1 + 16];
    }
    if (flags & 1) {
      v00 = fmaxf(v00, 0.0f); v01 = fmaxf(v01, 0.0f);
      v10 = fmaxf(v10, 0.0f); v11 = fmaxf(v11, 0.0f);
    }
    C[o0] = v00; C[o0 + 16] = v01;
    C[o1] = v10; C[o1 + 16] = v11;
  }
}

// ---------------- attention scores (f16 operands, padded, guard-free) --------
// out[h,q,k] = (qh.kh)*SCALE + mask[q,k]
__global__ void __launch_bounds__(128)
attn_score_kernel(const _Float16* __restrict__ qh, const _Float16* __restrict__ kh,
                  const float* __restrict__ mask, float* __restrict__ out, int b) {
  int h = blockIdx.z;
  int wave = threadIdx.x >> 5;
  int lane = threadIdx.x & 31;
  int row0 = (blockIdx.y * 4 + wave) * 32;
  int col0 = blockIdx.x * 32;
  int m = lane & 15, g = lane >> 4;
  const _Float16* Q  = qh + (size_t)(b * NHD + h) * NQP * HD;
  const _Float16* Kh = kh + (size_t)(b * NHD + h) * NQP * HD;
  const _Float16* ar0 = Q + (size_t)(row0 + m) * HD + (g << 4);
  const _Float16* ar1 = ar0 + (size_t)16 * HD;
  v16h af0 = cat8(ar0, ar0 + 8);
  v16h af1 = cat8(ar1, ar1 + 8);
  v16h bf0 = *(const v16h*)(Kh + (size_t)(col0 + m) * HD + (g << 4));
  v16h bf1 = *(const v16h*)(Kh + (size_t)(col0 + 16 + m) * HD + (g << 4));
  v8f a00 = {}, a01 = {}, a10 = {}, a11 = {};
  a00 = wmma_f16(af0, bf0, a00);
  a01 = wmma_f16(af0, bf1, a01);
  a10 = wmma_f16(af1, bf0, a10);
  a11 = wmma_f16(af1, bf1, a11);
  const float* Mb = mask + (size_t)b * NQP * NQP;
  float* O = out + (size_t)h * NQP * NQP;
  int cn = col0 + m;
#pragma unroll
  for (int r = 0; r < 8; ++r) {
    int cm0 = row0 + r + (g << 3);
    int cm1 = cm0 + 16;
    size_t o0 = (size_t)cm0 * NQP + cn;
    size_t o1 = (size_t)cm1 * NQP + cn;
    O[o0]      = a00[r] * SCALE + Mb[o0];
    O[o0 + 16] = a01[r] * SCALE + Mb[o0 + 16];
    O[o1]      = a10[r] * SCALE + Mb[o1];
    O[o1 + 16] = a11[r] * SCALE + Mb[o1 + 16];
  }
}

// ---------------- attn @ V : oh[h,q,hd] ; V pre-transposed f16 (b,h,hd,NQP) ---
__global__ void __launch_bounds__(128)
attn_av_kernel(const float* __restrict__ attn, const _Float16* __restrict__ vt,
               float* __restrict__ oh, int b) {
  int h = blockIdx.z;
  int wave = threadIdx.x >> 5;
  int lane = threadIdx.x & 31;
  int row0 = (blockIdx.y * 4 + wave) * 32;
  int m = lane & 15, g = lane >> 4;
  const float* A = attn + (size_t)h * NQP * NQP;
  const _Float16* V = vt + (size_t)(b * NHD + h) * HD * NQP;
  const float* arow0 = A + (size_t)(row0 + m) * NQP + (g << 4);
  const float* arow1 = arow0 + (size_t)16 * NQP;
  const _Float16* bcol0 = V + (size_t)m * NQP + (g << 4);
  const _Float16* bcol1 = bcol0 + (size_t)16 * NQP;
  v8f a00 = {}, a01 = {}, a10 = {}, a11 = {};
  for (int k0 = 0; k0 < NQP; k0 += 32) {
    v16h af0 = pack16(arow0 + k0, arow0 + k0 + 8);
    v16h af1 = pack16(arow1 + k0, arow1 + k0 + 8);
    v16h bf0 = *(const v16h*)(bcol0 + k0);
    v16h bf1 = *(const v16h*)(bcol1 + k0);
    a00 = wmma_f16(af0, bf0, a00);
    a01 = wmma_f16(af0, bf1, a01);
    a10 = wmma_f16(af1, bf0, a10);
    a11 = wmma_f16(af1, bf1, a11);
  }
  float* O = oh + (size_t)(b * NHD + h) * NQP * HD;
#pragma unroll
  for (int r = 0; r < 8; ++r) {
    int cm0 = row0 + r + (g << 3);
    int cm1 = cm0 + 16;
    O[(size_t)cm0 * HD + m]      = a00[r];
    O[(size_t)cm0 * HD + m + 16] = a01[r];
    O[(size_t)cm1 * HD + m]      = a10[r];
    O[(size_t)cm1 * HD + m + 16] = a11[r];
  }
}

// ---------------- one-time weight transpose+convert: W(KxN,f32) -> Wt(NxK,f16)
__global__ void wtrans_kernel(const float* __restrict__ W, _Float16* __restrict__ Wt,
                              int Kd, int Nd) {
  int idx = blockIdx.x * 256 + threadIdx.x;
  if (idx >= Kd * Nd) return;
  int n = idx % Nd, k = idx / Nd;
  Wt[(size_t)n * Kd + k] = (_Float16)W[idx];
}

// ---------------- row softmax (wave per row) ----------------
__global__ void __launch_bounds__(256)
softmax_rows_kernel(float* __restrict__ x, int rows, int n) {
  int wid = blockIdx.x * 8 + (threadIdx.x >> 5);
  int lane = threadIdx.x & 31;
  if (wid >= rows) return;
  float* p = x + (size_t)wid * n;
  float mx = -3.4e38f;
  for (int i = lane; i < n; i += 32) mx = fmaxf(mx, p[i]);
#pragma unroll
  for (int o = 16; o > 0; o >>= 1) mx = fmaxf(mx, __shfl_xor(mx, o, 32));
  float s = 0.0f;
  for (int i = lane; i < n; i += 32) { float e = __expf(p[i] - mx); p[i] = e; s += e; }
#pragma unroll
  for (int o = 16; o > 0; o >>= 1) s += __shfl_xor(s, o, 32);
  float inv = 1.0f / s;
  for (int i = lane; i < n; i += 32) p[i] *= inv;
}

// ---------------- fused (x [+resid]) -> LayerNorm, D=256, wave per row --------
__global__ void __launch_bounds__(256)
add_ln_kernel(const float* __restrict__ x, const float* __restrict__ resid,
              const float* __restrict__ gw, const float* __restrict__ bw,
              float* __restrict__ out, int rows) {
  int wid = blockIdx.x * 8 + (threadIdx.x >> 5);
  int lane = threadIdx.x & 31;
  if (wid >= rows) return;
  size_t base = (size_t)wid * DIM;
  float v[8];
  float s = 0.0f;
#pragma unroll
  for (int j = 0; j < 8; ++j) {
    int c = lane + 32 * j;
    float t = x[base + c];
    if (resid) t += resid[base + c];
    v[j] = t; s += t;
  }
#pragma unroll
  for (int o = 16; o > 0; o >>= 1) s += __shfl_xor(s, o, 32);
  float mean = s * (1.0f / DIM);
  float var = 0.0f;
#pragma unroll
  for (int j = 0; j < 8; ++j) { float d = v[j] - mean; var += d * d; }
#pragma unroll
  for (int o = 16; o > 0; o >>= 1) var += __shfl_xor(var, o, 32);
  float inv = rsqrtf(var * (1.0f / DIM) + 1e-5f);
#pragma unroll
  for (int j = 0; j < 8; ++j) {
    int c = lane + 32 * j;
    out[base + c] = (v[j] - mean) * inv * gw[c] + bw[c];
  }
}

// ---------------- elementwise ----------------
__global__ void add_kernel(const float* __restrict__ a, const float* __restrict__ b,
                           float* __restrict__ c, int n) {
  int i = blockIdx.x * 256 + threadIdx.x;
  if (i < n) c[i] = a[i] + b[i];
}
__global__ void fill_kernel(float* __restrict__ p, float v, int n) {
  int i = blockIdx.x * 256 + threadIdx.x;
  if (i < n) p[i] = v;
}

// ---------------- head pack (f16) ----------------
// (NQ*B, D) f32 -> (B, NH, NQP, HD) f16, zero pad rows
__global__ void pack_qk_kernel(const float* __restrict__ x, _Float16* __restrict__ y) {
  int idx = blockIdx.x * 256 + threadIdx.x;
  if (idx >= BB * NHD * NQP * HD) return;
  int hd = idx & 31;
  int t = idx >> 5;
  int q = t & (NQP - 1);
  t >>= 10;
  int h = t & (NHD - 1);
  int b = t >> 3;
  float v = 0.0f;
  if (q < NQ) v = x[((size_t)q * BB + b) * DIM + h * HD + hd];
  y[idx] = (_Float16)v;
}
// (NQ*B, D) f32 -> (B, NH, HD, NQP) f16 transposed, zero pad
__global__ void pack_vt_kernel(const float* __restrict__ x, _Float16* __restrict__ y) {
  int idx = blockIdx.x * 256 + threadIdx.x;
  if (idx >= BB * NHD * HD * NQP) return;
  int q = idx & (NQP - 1);
  int t = idx >> 10;
  int hd = t & 31;
  t >>= 5;
  int h = t & (NHD - 1);
  int b = t >> 3;
  float v = 0.0f;
  if (q < NQ) v = x[((size_t)q * BB + b) * DIM + h * HD + hd];
  y[idx] = (_Float16)v;
}
// (B, NH, NQP, HD) f32 -> (NQ*B, D)
__global__ void unpack_heads_kernel(const float* __restrict__ x, float* __restrict__ y) {
  int idx = blockIdx.x * 256 + threadIdx.x;
  if (idx >= NQ * BB * DIM) return;
  int d = idx & (DIM - 1); int i = idx >> 8;
  int b = i & (BB - 1); int q = i >> 2;
  int h = d >> 5; int hd = d & 31;
  y[idx] = x[(((size_t)(b * NHD + h)) * NQP + q) * HD + hd];
}

// ---------------- dense mask pipeline ----------------
__global__ void sigmoid_coords_kernel(const float* __restrict__ pc, float* __restrict__ rs) {
  int idx = blockIdx.x * 256 + threadIdx.x;
  if (idx >= BB * NQ * 4) return;
  int c = idx & 3; int r = idx >> 2;
  int q = r % NQ; int b = r / NQ;
  float x = pc[((size_t)q * BB + b) * 4 + c];
  rs[idx] = 1.0f / (1.0f + __expf(-x));
}

__global__ void giou_score_kernel(const float* __restrict__ rs, float* __restrict__ score) {
  size_t idx = (size_t)blockIdx.x * 256 + threadIdx.x;
  if (idx >= (size_t)BB * NQ * NQ) return;
  int k = idx % NQ; size_t t = idx / NQ;
  int q = t % NQ; int b = t / NQ;
  const float* b1 = rs + ((size_t)b * NQ + q) * 4;
  const float* b2 = rs + ((size_t)b * NQ + k) * 4;
  float ax1 = b1[0] - 0.5f * b1[2], ay1 = b1[1] - 0.5f * b1[3];
  float ax2 = b1[0] + 0.5f * b1[2], ay2 = b1[1] + 0.5f * b1[3];
  float bx1 = b2[0] - 0.5f * b2[2], by1 = b2[1] - 0.5f * b2[3];
  float bx2 = b2[0] + 0.5f * b2[2], by2 = b2[1] + 0.5f * b2[3];
  float a1 = (ax2 - ax1) * (ay2 - ay1);
  float a2 = (bx2 - bx1) * (by2 - by1);
  float iw = fminf(ax2, bx2) - fmaxf(ax1, bx1);
  float ih = fminf(ay2, by2) - fmaxf(ay1, by1);
  float inter = fmaxf(iw, 0.0f) * fmaxf(ih, 0.0f);
  float uni = a1 + a2 - inter;
  float iou = inter / uni;
  float ew = fmaxf(ax2, bx2) - fminf(ax1, bx1);
  float eh = fmaxf(ay2, by2) - fminf(ay1, by1);
  float ae = fmaxf(ew, 0.0f) * fmaxf(eh, 0.0f);
  float giou = iou - (ae - uni) / ae;
  score[idx] = 1.0f - giou;
}

__global__ void __launch_bounds__(256)
topk_mask_kernel(const float* __restrict__ score, float* __restrict__ mask) {
  __shared__ float s[NQ];
  __shared__ unsigned char flag[NQ];
  __shared__ float rmin[256];
  __shared__ int   rind[256];
  int t = threadIdx.x;
  int row = blockIdx.x;             // row = b*NQ + q
  int b = row / NQ, q = row % NQ;
  const float* p = score + (size_t)row * NQ;
  for (int i = t; i < NQ; i += 256) { s[i] = p[i]; flag[i] = 0; }
  __syncthreads();
  for (int it = 0; it < TOPK; ++it) {
    float lm = 3.4e38f; int li = -1;
    for (int i = t; i < NQ; i += 256)
      if (s[i] < lm) { lm = s[i]; li = i; }
    rmin[t] = lm; rind[t] = li;
    __syncthreads();
    for (int o = 128; o > 0; o >>= 1) {
      if (t < o && rmin[t + o] < rmin[t]) { rmin[t] = rmin[t + o]; rind[t] = rind[t + o]; }
      __syncthreads();
    }
    if (t == 0) { int w = rind[0]; if (w >= 0) { flag[w] = 1; s[w] = 3.4e38f; } }
    __syncthreads();
  }
  float* mrow = mask + ((size_t)b * NQP + q) * NQP;
  for (int i = t; i < NQ; i += 256) mrow[i] = flag[i] ? 1.0f : 0.0f;
}

// ---------------- deformable attention pieces ----------------
__global__ void __launch_bounds__(256)
aw_softmax_kernel(const float* __restrict__ raw, float* __restrict__ aw,
                  float* __restrict__ dec) {
  int wid = blockIdx.x * 8 + (threadIdx.x >> 5);
  int lane = threadIdx.x & 31;
  if (wid >= NQ * BB * NHD) return;
  int h = wid % NHD; int i = wid / NHD;
  int b = i & (BB - 1); int q = i >> 2;
  float v = raw[(size_t)i * DIM + h * 32 + lane];
  float mx = v;
#pragma unroll
  for (int o = 16; o > 0; o >>= 1) mx = fmaxf(mx, __shfl_xor(mx, o, 32));
  float e = __expf(v - mx);
  float s = e;
#pragma unroll
  for (int o = 16; o > 0; o >>= 1) s += __shfl_xor(s, o, 32);
  float r = e / s;
  aw[(size_t)i * DIM + h * 32 + lane] = r;
  dec[(((size_t)(b * NQ + q)) * NHD + h) * 32 + lane] = r;
}

DEV float samp(const float* __restrict__ val, int start, int b, int h,
               int Hl, int Wl, int yi, int xi, int lane) {
  if (xi < 0 || xi >= Wl || yi < 0 || yi >= Hl) return 0.0f;
  int s = start + yi * Wl + xi;
  return val[(((size_t)s * BB + b) * NHD + h) * HD + lane];
}

__global__ void __launch_bounds__(256)
deform_sample_kernel(const float* __restrict__ val, const float* __restrict__ offb,
                     const float* __restrict__ aw, const float* __restrict__ rs,
                     float* __restrict__ out) {
  int wid = blockIdx.x * 8 + (threadIdx.x >> 5);
  int lane = threadIdx.x & 31;
  if (wid >= NQ * BB * NHD) return;
  int h = wid % NHD; int i = wid / NHD;
  int b = i & (BB - 1); int q = i >> 2;
  const float* r = rs + ((size_t)b * NQ + q) * 4;
  float r0 = r[0], r1 = r[1], r2 = r[2], r3 = r[3];
  const int HL[4] = {64, 32, 16, 8};
  float acc = 0.0f;
  int start = 0;
#pragma unroll
  for (int l = 0; l < LVN; ++l) {
    int Hl = HL[l], Wl = HL[l];
#pragma unroll
    for (int p = 0; p < PN; ++p) {
      size_t ob = (size_t)i * (NHD * LVN * PN * 2) + (((h * LVN + l) * PN + p) * 2);
      float ox = offb[ob], oy = offb[ob + 1];
      float lx = r0 + ox * (1.0f / PN) * r2 * 0.5f;
      float ly = r1 + oy * (1.0f / PN) * r3 * 0.5f;
      float x = lx * Wl - 0.5f;
      float y = ly * Hl - 0.5f;
      float x0f = floorf(x), y0f = floorf(y);
      float fx = x - x0f, fy = y - y0f;
      int x0 = (int)x0f, y0 = (int)y0f;
      float a = aw[(size_t)i * DIM + h * 32 + l * PN + p];
      float v00 = samp(val, start, b, h, Hl, Wl, y0,     x0,     lane);
      float v01 = samp(val, start, b, h, Hl, Wl, y0,     x0 + 1, lane);
      float v10 = samp(val, start, b, h, Hl, Wl, y0 + 1, x0,     lane);
      float v11 = samp(val, start, b, h, Hl, Wl, y0 + 1, x0 + 1, lane);
      acc += a * (v00 * (1.0f - fx) * (1.0f - fy) + v01 * fx * (1.0f - fy) +
                  v10 * (1.0f - fx) * fy          + v11 * fx * fy);
    }
    start += Hl * Wl;
  }
  out[(size_t)i * DIM + h * 32 + lane] = acc;
}

// ---------------- host orchestration ----------------
static inline dim3 gemm_grid(int M, int N) {
  return dim3(N / 32, (M + 127) / 128, 1);
}

// per-layer half offsets inside the transposed-weight arena
constexpr size_t HW_Q   = 0;
constexpr size_t HW_K   = 65536;
constexpr size_t HW_V   = 131072;
constexpr size_t HW_O   = 196608;
constexpr size_t HW_VAL = 262144;
constexpr size_t HW_OUT = 327680;
constexpr size_t HW_OFF = 393216;   // 512*256
constexpr size_t HW_AW  = 524288;   // 256*256
constexpr size_t HW_W1  = 589824;   // 2048*256
constexpr size_t HW_W2  = 1114112;  // 256*2048
constexpr size_t HW_LSTRIDE = 1638400;

extern "C" void kernel_launch(void* const* d_in, const int* in_sizes, int n_in,
                              void* d_out, int out_size, void* d_ws, size_t ws_size,
                              hipStream_t stream) {
  const float* tgt        = (const float*)d_in[0];
  const float* memoryP    = (const float*)d_in[1];
  const float* query_pos  = (const float*)d_in[2];
  const float* pos_centers= (const float*)d_in[3];
  const float* Wq  = (const float*)d_in[4];
  const float* bq  = (const float*)d_in[5];
  const float* Wk  = (const float*)d_in[6];
  const float* bk  = (const float*)d_in[7];
  const float* Wv  = (const float*)d_in[8];
  const float* bv  = (const float*)d_in[9];
  const float* Wo  = (const float*)d_in[10];
  const float* bo  = (const float*)d_in[11];
  const float* Woff= (const float*)d_in[12];
  const float* boff= (const float*)d_in[13];
  const float* Waw = (const float*)d_in[14];
  const float* baw = (const float*)d_in[15];
  const float* Wval= (const float*)d_in[16];
  const float* bval= (const float*)d_in[17];
  const float* Wout= (const float*)d_in[18];
  const float* bout= (const float*)d_in[19];
  const float* W1  = (const float*)d_in[20];
  const float* b1  = (const float*)d_in[21];
  const float* W2  = (const float*)d_in[22];
  const float* b2  = (const float*)d_in[23];
  const float* lnw = (const float*)d_in[24];
  const float* lnb = (const float*)d_in[25];
  const float* gn  = (const float*)d_in[26];
  const float* gnb = (const float*)d_in[27];

  float* out0    = (float*)d_out;                    // (NQ,B,D)
  float* outpc   = out0 + (size_t)NQ * BB * DIM;     // (NQ,B,4)
  float* outattn = outpc + (size_t)NQ * BB * 4;      // (B,NQ,NH,32)

  // ---- workspace carving: f16 arena first (alignment), then f32 ----
  _Float16* wsh = (_Float16*)d_ws;
  size_t hoff = 0;
  auto alloch = [&](size_t n) { _Float16* p = wsh + hoff; hoff += n; return p; };
  _Float16* WT   = alloch(HW_LSTRIDE * NL);            // transposed f16 weights
  _Float16* qh   = alloch((size_t)BB * NHD * NQP * HD);
  _Float16* kh   = alloch((size_t)BB * NHD * NQP * HD);
  _Float16* vt   = alloch((size_t)BB * NHD * HD * NQP);
  // hoff is a multiple of 16 halves -> f32 region stays 32B aligned
  float* wsf = (float*)(wsh + hoff);
  size_t foff = 0;
  auto allocf = [&](size_t n) { float* p = wsf + foff; foff += n; return p; };
  float* mask    = allocf((size_t)BB * NQP * NQP);
  float* attnbuf = allocf((size_t)NHD * NQP * NQP);    // per-b scores / giou scratch
  float* tgt_cur = allocf((size_t)NQ * BB * DIM);
  float* qbuf    = allocf((size_t)NQ * BB * DIM);
  float* tmp0    = allocf((size_t)NQ * BB * DFF);
  float* oheads  = allocf((size_t)BB * NHD * NQP * HD);
  float* valbuf  = allocf((size_t)STOT * BB * DIM);
  float* offbuf  = allocf((size_t)NQ * BB * NHD * LVN * PN * 2);
  float* awbuf   = allocf((size_t)NQ * BB * DIM);
  float* dattn   = allocf((size_t)NQ * BB * DIM);
  float* t2buf   = allocf((size_t)NQ * BB * DIM);
  float* refsig  = allocf((size_t)BB * NQ * 4);
  (void)ws_size; (void)in_sizes; (void)n_in; (void)out_size;

  const int NE = NQ * BB * DIM;           // 1,024,000
  const int EW_BLOCKS = (NE + 255) / 256;
  const int NPACK = BB * NHD * NQP * HD;  // 1,048,576

  // ---- one-time weight transpose/convert to f16 (N-major) ----
  for (int l = 0; l < NL; ++l) {
    _Float16* base = WT + HW_LSTRIDE * (size_t)l;
    auto tr = [&](const float* W, size_t hofs, int Kd, int Nd) {
      wtrans_kernel<<<(Kd * Nd + 255) / 256, 256, 0, stream>>>(W, base + hofs, Kd, Nd);
    };
    tr(Wq  + (size_t)l * DIM * DIM, HW_Q,   DIM, DIM);
    tr(Wk  + (size_t)l * DIM * DIM, HW_K,   DIM, DIM);
    tr(Wv  + (size_t)l * DIM * DIM, HW_V,   DIM, DIM);
    tr(Wo  + (size_t)l * DIM * DIM, HW_O,   DIM, DIM);
    tr(Wval+ (size_t)l * DIM * DIM, HW_VAL, DIM, DIM);
    tr(Wout+ (size_t)l * DIM * DIM, HW_OUT, DIM, DIM);
    tr(Woff+ (size_t)l * DIM * 512, HW_OFF, DIM, 512);
    tr(Waw + (size_t)l * DIM * DIM, HW_AW,  DIM, DIM);
    tr(W1  + (size_t)l * DIM * DFF, HW_W1,  DIM, DFF);
    tr(W2  + (size_t)l * DFF * DIM, HW_W2,  DFF, DIM);
  }

  // ---- pos_centers passthrough output ----
  hipMemcpyAsync(outpc, pos_centers, (size_t)NQ * BB * 4 * sizeof(float),
                 hipMemcpyDeviceToDevice, stream);

  // ---- dense attention mask (padded with NEGBIG) ----
  sigmoid_coords_kernel<<<(BB * NQ * 4 + 255) / 256, 256, 0, stream>>>(pos_centers, refsig);
  giou_score_kernel<<<(BB * NQ * NQ + 255) / 256, 256, 0, stream>>>(refsig, attnbuf);
  fill_kernel<<<(BB * NQP * NQP + 255) / 256, 256, 0, stream>>>(mask, NEGBIG, BB * NQP * NQP);
  topk_mask_kernel<<<BB * NQ, 256, 0, stream>>>(attnbuf, mask);

  // ---- tgt_cur = tgt ----
  hipMemcpyAsync(tgt_cur, tgt, (size_t)NE * sizeof(float),
                 hipMemcpyDeviceToDevice, stream);

  for (int l = 0; l < NL; ++l) {
    const _Float16* base = WT + HW_LSTRIDE * (size_t)l;
    const float* bq_l  = bq  + (size_t)l * DIM;
    const float* bk_l  = bk  + (size_t)l * DIM;
    const float* bv_l  = bv  + (size_t)l * DIM;
    const float* bo_l  = bo  + (size_t)l * DIM;
    const float* bval_l= bval+ (size_t)l * DIM;
    const float* bout_l= bout+ (size_t)l * DIM;
    const float* baw_l = baw + (size_t)l * 256;
    const float* boff_l= boff+ (size_t)l * 512;
    const float* b1_l  = b1  + (size_t)l * DFF;
    const float* b2_l  = b2  + (size_t)l * DIM;
    const float* ln0 = lnw + ((size_t)l * 3 + 0) * DIM;
    const float* ln1 = lnw + ((size_t)l * 3 + 1) * DIM;
    const float* ln2 = lnw + ((size_t)l * 3 + 2) * DIM;
    const float* lb0 = lnb + ((size_t)l * 3 + 0) * DIM;
    const float* lb1 = lnb + ((size_t)l * 3 + 1) * DIM;
    const float* lb2 = lnb + ((size_t)l * 3 + 2) * DIM;

    // ===== self attention =====
    add_kernel<<<EW_BLOCKS, 256, 0, stream>>>(tgt_cur, query_pos, qbuf, NE);
    gemm_nn<DIM, DIM><<<gemm_grid(NQ * BB, DIM), 128, 0, stream>>>(
        qbuf, base + HW_Q, bq_l, nullptr, tmp0, NQ * BB, 0);
    pack_qk_kernel<<<(NPACK + 255) / 256, 256, 0, stream>>>(tmp0, qh);
    gemm_nn<DIM, DIM><<<gemm_grid(NQ * BB, DIM), 128, 0, stream>>>(
        qbuf, base + HW_K, bk_l, nullptr, tmp0, NQ * BB, 0);
    pack_qk_kernel<<<(NPACK + 255) / 256, 256, 0, stream>>>(tmp0, kh);
    gemm_nn<DIM, DIM><<<gemm_grid(NQ * BB, DIM), 128, 0, stream>>>(
        tgt_cur, base + HW_V, bv_l, nullptr, tmp0, NQ * BB, 0);
    pack_vt_kernel<<<(NPACK + 255) / 256, 256, 0, stream>>>(tmp0, vt);

    for (int b = 0; b < BB; ++b) {
      attn_score_kernel<<<dim3(NQP / 32, NQP / 128, NHD), 128, 0, stream>>>(
          qh, kh, mask, attnbuf, b);
      softmax_rows_kernel<<<(NHD * NQP + 7) / 8, 256, 0, stream>>>(
          attnbuf, NHD * NQP, NQP);
      attn_av_kernel<<<dim3(1, NQP / 128, NHD), 128, 0, stream>>>(
          attnbuf, vt, oheads, b);
    }
    unpack_heads_kernel<<<EW_BLOCKS, 256, 0, stream>>>(oheads, t2buf);
    gemm_nn<DIM, DIM><<<gemm_grid(NQ * BB, DIM), 128, 0, stream>>>(
        t2buf, base + HW_O, bo_l, nullptr, tmp0, NQ * BB, 0);
    add_ln_kernel<<<(NQ * BB + 7) / 8, 256, 0, stream>>>(
        tmp0, tgt_cur, ln0, lb0, tgt_cur, NQ * BB);

    // ===== deformable attention =====
    add_kernel<<<EW_BLOCKS, 256, 0, stream>>>(tgt_cur, query_pos, qbuf, NE);
    gemm_nn<512, DIM><<<gemm_grid(NQ * BB, 512), 128, 0, stream>>>(
        qbuf, base + HW_OFF, boff_l, nullptr, offbuf, NQ * BB, 0);
    gemm_nn<DIM, DIM><<<gemm_grid(NQ * BB, DIM), 128, 0, stream>>>(
        qbuf, base + HW_AW, baw_l, nullptr, tmp0, NQ * BB, 0);
    aw_softmax_kernel<<<(NQ * BB * NHD + 7) / 8, 256, 0, stream>>>(tmp0, awbuf, outattn);
    gemm_nn<DIM, DIM><<<gemm_grid(STOT * BB, DIM), 128, 0, stream>>>(
        memoryP, base + HW_VAL, bval_l, nullptr, valbuf, STOT * BB, 0);
    deform_sample_kernel<<<(NQ * BB * NHD + 7) / 8, 256, 0, stream>>>(
        valbuf, offbuf, awbuf, refsig, dattn);
    gemm_nn<DIM, DIM><<<gemm_grid(NQ * BB, DIM), 128, 0, stream>>>(
        dattn, base + HW_OUT, bout_l, tgt_cur, tgt_cur, NQ * BB, 0);

    // ===== FFN =====
    add_ln_kernel<<<(NQ * BB + 7) / 8, 256, 0, stream>>>(
        tgt_cur, nullptr, ln1, lb1, t2buf, NQ * BB);
    gemm_nn<DFF, DIM><<<gemm_grid(NQ * BB, DFF), 128, 0, stream>>>(
        t2buf, base + HW_W1, b1_l, nullptr, tmp0, NQ * BB, 1 /*relu*/);
    gemm_nn<DIM, DFF><<<gemm_grid(NQ * BB, DIM), 128, 0, stream>>>(
        tmp0, base + HW_W2, b2_l, nullptr, t2buf, NQ * BB, 0);
    add_ln_kernel<<<(NQ * BB + 7) / 8, 256, 0, stream>>>(
        t2buf, tgt_cur, ln2, lb2, tgt_cur, NQ * BB);
  }

  // ===== final norm =====
  add_ln_kernel<<<(NQ * BB + 7) / 8, 256, 0, stream>>>(
      tgt_cur, nullptr, gn, gnb, out0, NQ * BB);
}
